// RFS5_72662256714370
// MI455X (gfx1250) — matrix operand order repaired and checked by reading, not compiled, not run on hardware
//
#include <hip/hip_runtime.h>
#include <stdint.h>

// ---------------------------------------------------------------------------
// S5 diagonal SSM for MI455X (gfx1250, wave32).
//   y = (scan(exp(Lambda*dt), u @ W1)) @ W2   with complex arithmetic folded
// into real bf16 WMMA GEMMs (K,N doubled; re/im interleaved so GEMM1 reads
// u's native layout and GEMM2 writes the output's native layout).
//
// Pipeline: prep_weights -> lambda -> convert(u->bf16) -> GEMM1(WMMA) ->
//           scan_local -> scan_carry -> scan_fixup(writes bf16 xs) ->
//           GEMM2(WMMA).
// GEMM K-loop is software-pipelined two deep in registers so each step's
// WMMAs overlap the next step's A/B fragment loads (waits become
// loadcnt<=8 instead of 0).
// ---------------------------------------------------------------------------

typedef __attribute__((ext_vector_type(16))) __bf16 v16bf;
typedef __attribute__((ext_vector_type(8)))  float  v8f;

namespace s5 {
constexpr int BATCH = 8, SEQ = 8192, HDIM = 512, PDIM = 512;
constexpr int KD = 2 * HDIM;          // 1024 = GEMM K (also 2*PDIM for GEMM2)
constexpr int ND = 2 * PDIM;          // 1024 = GEMM N
constexpr int MROWS = BATCH * SEQ;    // 65536 GEMM M
constexpr int NCHUNK = 8, TCHUNK = SEQ / NCHUNK;   // 8 chunks of 1024

// Workspace layout (bytes).
constexpr size_t SZ_W    = (size_t)KD * ND * 2;                 // 2 MiB bf16
constexpr size_t OFF_W1  = 0;
constexpr size_t OFF_W2  = OFF_W1 + SZ_W;
constexpr size_t OFF_LAM = OFF_W2 + SZ_W;                       // P float2
constexpr size_t OFF_LAMT= OFF_LAM + (size_t)PDIM * 8;
constexpr size_t OFF_FIN = OFF_LAMT + (size_t)PDIM * 8;         // B*NC*P float2
constexpr size_t SZ_FIN  = (size_t)BATCH * NCHUNK * PDIM * 8;
constexpr size_t OFF_CAR = OFF_FIN + SZ_FIN;
constexpr size_t OFF_BU  = (OFF_CAR + SZ_FIN + 255) & ~(size_t)255;
constexpr size_t SZ_BU   = (size_t)MROWS * ND * 4;              // 256 MiB f32
// u_bf16 region is dead once GEMM1 completes; reused for xs_bf16.
constexpr size_t OFF_ABF = OFF_BU + SZ_BU;                      // 128 MiB bf16
}
using namespace s5;

__device__ __forceinline__ unsigned short f2bf(float f) {
  union { __bf16 h; unsigned short u; } cv;
  cv.h = (__bf16)f;                       // fptrunc, round-to-nearest-even
  return cv.u;
}

__device__ __forceinline__ unsigned int pk_bf16(float a, float b) {
  union { __bf16 h[2]; unsigned int u; } cv;
  cv.h[0] = (__bf16)a;
  cv.h[1] = (__bf16)b;
  return cv.u;
}

// Swizzled B-fragment weight layout: per (32K x 16N) tile, 512 bf16 stored as
// lane*16 + j, where (lane, j) follow the CDNA5 dense 16-bit B-matrix layout:
// lanes 0-15 hold N=lane / K=0..7,16..23, lanes 16-31 hold K=8..15,24..31.
__device__ __forceinline__ size_t wswz(int k, int n) {
  int kb = k >> 5, nb = n >> 4, kl = k & 31, nl = n & 15;
  int lane = nl + 16 * ((kl >> 3) & 1);
  int j    = (kl & 7) + 8 * (kl >> 4);
  return (((size_t)kb * (ND >> 4) + nb) << 9) + lane * 16 + j;
}

// --------------------------- weight preparation ----------------------------
// W1 (2H x 2P): k=2h+c, n=2p+d.  Re col: [Br, -Bi], Im col: [Bi, Br].
// W2 (2P x 2H): k=2p+d, n=2h+c.  Re col: [Cr, -Ci], Im col: [Ci, Cr].
__global__ void s5_prep_weights(const float* __restrict__ Bm,
                                const float* __restrict__ Cm,
                                unsigned short* __restrict__ W1,
                                unsigned short* __restrict__ W2) {
  int tid = blockIdx.x * blockDim.x + threadIdx.x;
  const int total = KD * ND;
  if (tid < total) {
    int k = tid >> 10, n = tid & 1023;
    int h = k >> 1, c = k & 1, p = n >> 1, d = n & 1;
    float br = Bm[((size_t)p * HDIM + h) * 2 + 0];
    float bi = Bm[((size_t)p * HDIM + h) * 2 + 1];
    float v = (d == 0) ? (c == 0 ? br : -bi) : (c == 0 ? bi : br);
    W1[wswz(k, n)] = f2bf(v);
  } else if (tid < 2 * total) {
    int idx = tid - total;
    int k = idx >> 10, n = idx & 1023;
    int p = k >> 1, d = k & 1, h = n >> 1, c = n & 1;
    float cr = Cm[((size_t)h * PDIM + p) * 2 + 0];
    float ci = Cm[((size_t)h * PDIM + p) * 2 + 1];
    float v = (c == 0) ? (d == 0 ? cr : -ci) : (d == 0 ? ci : cr);
    W2[wswz(k, n)] = f2bf(v);
  }
}

// ------------------------------ discretization -----------------------------
__global__ void s5_lambda(const float* __restrict__ Lam,
                          const float* __restrict__ log_step,
                          float* __restrict__ lam, float* __restrict__ lamT) {
  int p = blockIdx.x * blockDim.x + threadIdx.x;
  if (p >= PDIM) return;
  float s  = expf(log_step[p]);
  float ar = Lam[2 * p], ai = Lam[2 * p + 1];
  float m = expf(ar * s), th = ai * s;
  lam[2 * p] = m * cosf(th);  lam[2 * p + 1] = m * sinf(th);
  // chunk aggregate A-product = Lambda_bar^TCHUNK, computed exactly
  float mT = expf(ar * s * (float)TCHUNK), thT = ai * s * (float)TCHUNK;
  lamT[2 * p] = mT * cosf(thT);  lamT[2 * p + 1] = mT * sinf(thT);
}

// ------------------------ f32 -> bf16 stream convert -----------------------
// Each thread converts 8 consecutive f32 into one packed uint4 of 8 bf16.
__global__ __launch_bounds__(256) void s5_f32_to_bf16(
    const float* __restrict__ src, unsigned short* __restrict__ dst,
    long long n8) {
  long long i = (long long)blockIdx.x * blockDim.x + threadIdx.x;
  if (i >= n8) return;
  const float4* s = (const float4*)src;
  float4 f0 = s[2 * i], f1 = s[2 * i + 1];
  uint4 o;
  o.x = pk_bf16(f0.x, f0.y);
  o.y = pk_bf16(f0.z, f0.w);
  o.z = pk_bf16(f1.x, f1.y);
  o.w = pk_bf16(f1.z, f1.w);
  ((uint4*)dst)[i] = o;
}

// ------------------------------- WMMA GEMM ---------------------------------
// C[M x 1024] = A_bf16[M x 1024] * W_bf16swz[1024 x 1024], f32 accumulate.
// Block: 256 thr = 8 waves (2 M-sub x 4 N-sub), tile 32(M) x 256(N); wave
// tile 16 x 64 (4 WMMA accumulators). Two-deep register pipeline: fragments
// for K-step k+1 load while step k's WMMAs execute.
__global__ __launch_bounds__(256) void s5_gemm_bf16(
    const unsigned short* __restrict__ Abf,
    const unsigned short* __restrict__ W, float* __restrict__ C) {
  const int tid  = threadIdx.x;
  const int lane = tid & 31, wave = tid >> 5;
  const int msub = wave & 1, nsub = wave >> 1;
  const int mbase = blockIdx.x * 32;
  const int nbase = blockIdx.y * 256 + nsub * 64;
  const int hi = lane >> 4;                // K-half select within fragment

  const int arow = mbase + msub * 16 + (lane & 15);
  const unsigned short* aptr = Abf + (size_t)arow * KD + (hi ? 8 : 0);
  const int nt0 = nbase >> 4;

  union F { v16bf v; uint4 q[2]; };

  auto loadA = [&](F& f, int kb) {
    // A fragment (16x32 bf16): lanes<16 take K 0-7 & 16-23 of row lane,
    // lanes>=16 take K 8-15 & 24-31 of row lane-16.
    const unsigned short* ak = aptr + kb * 32;
    f.q[0] = *(const uint4*)(ak);
    f.q[1] = *(const uint4*)(ak + 16);
  };
  auto loadB = [&](F& f0, F& f1, F& f2, F& f3, int kb) {
    // Four B fragments, 32 contiguous bytes per lane, L2-resident weights.
    const unsigned short* wb =
        W + (((size_t)kb * (ND >> 4) + nt0) << 9) + lane * 16;
    f0.q[0] = *(const uint4*)(wb + 0);    f0.q[1] = *(const uint4*)(wb + 8);
    f1.q[0] = *(const uint4*)(wb + 512);  f1.q[1] = *(const uint4*)(wb + 520);
    f2.q[0] = *(const uint4*)(wb + 1024); f2.q[1] = *(const uint4*)(wb + 1032);
    f3.q[0] = *(const uint4*)(wb + 1536); f3.q[1] = *(const uint4*)(wb + 1544);
  };

  v8f acc0 = {}, acc1 = {}, acc2 = {}, acc3 = {};
  constexpr int NK = KD / 32;              // 32 K-steps

  F aA, aB, bA0, bA1, bA2, bA3, bB0, bB1, bB2, bB3;
  loadA(aA, 0);
  loadB(bA0, bA1, bA2, bA3, 0);

  for (int kb = 0; kb < NK; kb += 2) {
    // prefetch step kb+1 into the B set while set A's WMMAs run
    loadA(aB, kb + 1);
    loadB(bB0, bB1, bB2, bB3, kb + 1);
    acc0 = __builtin_amdgcn_wmma_f32_16x16x32_bf16(false, aA.v, false, bA0.v,
                                                   (short)0, acc0, false, false);
    acc1 = __builtin_amdgcn_wmma_f32_16x16x32_bf16(false, aA.v, false, bA1.v,
                                                   (short)0, acc1, false, false);
    acc2 = __builtin_amdgcn_wmma_f32_16x16x32_bf16(false, aA.v, false, bA2.v,
                                                   (short)0, acc2, false, false);
    acc3 = __builtin_amdgcn_wmma_f32_16x16x32_bf16(false, aA.v, false, bA3.v,
                                                   (short)0, acc3, false, false);

    // prefetch step kb+2 into the A set (clamped: last iter's load is dead
    // but stays inside the buffers)
    const int k2 = (kb + 2 < NK) ? kb + 2 : 0;
    loadA(aA, k2);
    loadB(bA0, bA1, bA2, bA3, k2);
    acc0 = __builtin_amdgcn_wmma_f32_16x16x32_bf16(false, aB.v, false, bB0.v,
                                                   (short)0, acc0, false, false);
    acc1 = __builtin_amdgcn_wmma_f32_16x16x32_bf16(false, aB.v, false, bB1.v,
                                                   (short)0, acc1, false, false);
    acc2 = __builtin_amdgcn_wmma_f32_16x16x32_bf16(false, aB.v, false, bB2.v,
                                                   (short)0, acc2, false, false);
    acc3 = __builtin_amdgcn_wmma_f32_16x16x32_bf16(false, aB.v, false, bB3.v,
                                                   (short)0, acc3, false, false);
  }

  // D layout: VGPR r -> rows (hi*8 + r), col = lane&15 within 16-wide subtile.
  float* cp = C + (size_t)(mbase + msub * 16 + hi * 8) * ND + nbase + (lane & 15);
#pragma unroll
  for (int r = 0; r < 8; ++r) {
    cp[(size_t)r * ND + 0]  = acc0[r];
    cp[(size_t)r * ND + 16] = acc1[r];
    cp[(size_t)r * ND + 32] = acc2[r];
    cp[(size_t)r * ND + 48] = acc3[r];
  }
}

// ----------------------------- chunked scan --------------------------------
// Pass A: in-place local scan (x0=0) per (batch, chunk); write chunk finals.
__global__ __launch_bounds__(256) void s5_scan_local(
    float* __restrict__ Bu, const float* __restrict__ lam,
    float* __restrict__ finals) {
  const int lane = threadIdx.x & 31, wave = threadIdx.x >> 5;
  const int gw = blockIdx.x * 8 + wave;            // 0..1023
  const int b = gw >> 7, rem = gw & 127;
  const int chunk = rem >> 4, pg = rem & 15;
  const int p = (pg << 5) + lane;
  const float ar = lam[2 * p], ai = lam[2 * p + 1];
  float2* buf = (float2*)Bu + ((size_t)b * SEQ + (size_t)chunk * TCHUNK) * PDIM + p;
  float xr = 0.f, xi = 0.f;
  for (int t = 0; t < TCHUNK; ++t) {
    float2 v = buf[(size_t)t * PDIM];
    float nr = ar * xr - ai * xi + v.x;
    float ni = ar * xi + ai * xr + v.y;
    xr = nr; xi = ni;
    buf[(size_t)t * PDIM] = make_float2(xr, xi);
  }
  ((float2*)finals)[((size_t)b * NCHUNK + chunk) * PDIM + p] = make_float2(xr, xi);
}

// Pass B: sequential combine of the 8 chunk finals -> per-chunk carries.
__global__ void s5_scan_carry(const float* __restrict__ lamT,
                              const float* __restrict__ finals,
                              float* __restrict__ carry) {
  int tid = blockIdx.x * blockDim.x + threadIdx.x;
  if (tid >= BATCH * PDIM) return;
  int b = tid >> 9, p = tid & 511;
  float Ar = lamT[2 * p], Ai = lamT[2 * p + 1];
  float cr = 0.f, ci = 0.f;
  for (int c = 0; c < NCHUNK; ++c) {
    ((float2*)carry)[((size_t)b * NCHUNK + c) * PDIM + p] = make_float2(cr, ci);
    float2 f = ((const float2*)finals)[((size_t)b * NCHUNK + c) * PDIM + p];
    float nr = Ar * cr - Ai * ci + f.x;
    float ni = Ar * ci + Ai * cr + f.y;
    cr = nr; ci = ni;
  }
}

// Pass C: xs[t] = local[t] + Lambda_bar^(t+1)*carry, emitted as packed bf16
// in the (M, 2P) operand layout GEMM2 consumes.
__global__ __launch_bounds__(256) void s5_scan_fixup(
    const float* __restrict__ Bu, const float* __restrict__ lam,
    const float* __restrict__ carry, unsigned short* __restrict__ Xbf) {
  const int lane = threadIdx.x & 31, wave = threadIdx.x >> 5;
  const int gw = blockIdx.x * 8 + wave;
  const int b = gw >> 7, rem = gw & 127;
  const int chunk = rem >> 4, pg = rem & 15;
  const int p = (pg << 5) + lane;
  float2 cv = ((const float2*)carry)[((size_t)b * NCHUNK + chunk) * PDIM + p];
  const float ar = lam[2 * p], ai = lam[2 * p + 1];
  const float2* buf =
      (const float2*)Bu + ((size_t)b * SEQ + (size_t)chunk * TCHUNK) * PDIM + p;
  unsigned int* obf =
      (unsigned int*)Xbf + ((size_t)b * SEQ + (size_t)chunk * TCHUNK) * PDIM + p;
  float pr = ar, pi = ai;                          // Lambda_bar^1
  for (int t = 0; t < TCHUNK; ++t) {
    float2 v = buf[(size_t)t * PDIM];
    float vr = v.x + pr * cv.x - pi * cv.y;
    float vi = v.y + pr * cv.y + pi * cv.x;
    obf[(size_t)t * PDIM] = pk_bf16(vr, vi);
    float npr = pr * ar - pi * ai;
    float npi = pr * ai + pi * ar;
    pr = npr; pi = npi;
  }
}

// -------------------------------- launcher ---------------------------------
extern "C" void kernel_launch(void* const* d_in, const int* in_sizes, int n_in,
                              void* d_out, int out_size, void* d_ws,
                              size_t ws_size, hipStream_t stream) {
  (void)in_sizes; (void)n_in; (void)out_size; (void)ws_size;
  const float* u        = (const float*)d_in[0];  // (8, 8192, 512, 2)
  const float* Lambda   = (const float*)d_in[1];  // (512, 2)
  const float* Bm       = (const float*)d_in[2];  // (512, 512, 2)
  const float* Cm       = (const float*)d_in[3];  // (512, 512, 2)
  const float* log_step = (const float*)d_in[4];  // (512, 1)
  float* out = (float*)d_out;                     // (8, 8192, 512, 2)

  char* ws = (char*)d_ws;
  unsigned short* W1 = (unsigned short*)(ws + OFF_W1);
  unsigned short* W2 = (unsigned short*)(ws + OFF_W2);
  float* lam  = (float*)(ws + OFF_LAM);
  float* lamT = (float*)(ws + OFF_LAMT);
  float* fin  = (float*)(ws + OFF_FIN);
  float* car  = (float*)(ws + OFF_CAR);
  float* Bu   = (float*)(ws + OFF_BU);            // (B*L, 2P) f32
  unsigned short* Abf = (unsigned short*)(ws + OFF_ABF);  // bf16 A operand

  s5_prep_weights<<<(2 * KD * ND + 255) / 256, 256, 0, stream>>>(Bm, Cm, W1, W2);
  s5_lambda<<<(PDIM + 255) / 256, 256, 0, stream>>>(Lambda, log_step, lam, lamT);

  // u -> bf16 (native layout already matches the (M, 2H) GEMM1 operand)
  const long long n8 = (long long)MROWS * KD / 8;
  s5_f32_to_bf16<<<(int)((n8 + 255) / 256), 256, 0, stream>>>(u, Abf, n8);

  dim3 gg(MROWS / 32, ND / 256);                  // 2048 x 4
  s5_gemm_bf16<<<gg, 256, 0, stream>>>(Abf, W1, Bu);

  s5_scan_local<<<128, 256, 0, stream>>>(Bu, lam, fin);
  s5_scan_carry<<<(BATCH * PDIM + 255) / 256, 256, 0, stream>>>(lamT, fin, car);
  // fixup overwrites the (now dead) u_bf16 region with xs_bf16
  s5_scan_fixup<<<128, 256, 0, stream>>>(Bu, lam, car, Abf);

  s5_gemm_bf16<<<gg, 256, 0, stream>>>(Abf, W2, out);
}
